// CNNPathFinder32Net_33973191311915
// MI455X (gfx1250) — matrix operand order repaired and checked by reading, not compiled, not run on hardware
//
#include <hip/hip_runtime.h>
#include <hip/hip_bf16.h>
#include <math.h>

typedef _Float16 h16 __attribute__((ext_vector_type(16)));
typedef float    f8  __attribute__((ext_vector_type(8)));

#define WMMA_F16(a, b, c) \
  __builtin_amdgcn_wmma_f32_16x16x32_f16(false, (a), false, (b), (short)0, (c), false, false)

// A-fragment (16x32 f16): lane holds row m = lane&15.
// k(e) = (e>>3)*16 + laneHi*8 + (e&7)  ->  two 16B loads at +laneHi*8 and +16+laneHi*8.
__device__ __forceinline__ h16 ldfragA(const _Float16* row, int laneHi) {
  union { h16 h; uint4 u[2]; } f;
  f.u[0] = *(const uint4*)(row + laneHi * 8);
  f.u[1] = *(const uint4*)(row + 16 + laneHi * 8);
  return f.h;
}
// B-fragment (32x16 f16): lane holds column n = lane&15, k(e) = (e>>3)*16 + laneHi*8 + (e&7).
__device__ __forceinline__ h16 ldfragB(const _Float16* col, int laneHi) {
  union { h16 h; uint4 u[2]; } f;
  const _Float16* p = col + laneHi * 8;
  f.u[0] = *(const uint4*)(p);
  f.u[1] = *(const uint4*)(p + 16);
  return f.h;
}

// order-preserving float<->uint encoding so we can max-pool with atomicMax(u32)
__device__ __forceinline__ unsigned int fenc(float x) {
  unsigned int b = __float_as_uint(x);
  return (b & 0x80000000u) ? ~b : (b | 0x80000000u);
}
__device__ __forceinline__ float fdec(unsigned int k) {
  unsigned int b = (k & 0x80000000u) ? (k & 0x7fffffffu) : ~k;
  return __uint_as_float(b);
}
#define ENC_NEGINF 0x007FFFFFu

// ---------------------------------------------------------------------------
// Weight conversion/reorder: conv weights -> [tap][oc][ic] f16, FC -> f16 copy
// ---------------------------------------------------------------------------
__global__ __launch_bounds__(256) void k_cvt(
    const float* __restrict__ w2, const float* __restrict__ w3, const float* __restrict__ w4,
    const float* __restrict__ cw1, const float* __restrict__ cw2,
    const float* __restrict__ ew1, const float* __restrict__ ew2,
    _Float16* o2, _Float16* o3, _Float16* o4,
    _Float16* oc1, _Float16* oc2, _Float16* oe1, _Float16* oe2)
{
  int d = blockIdx.x * 256 + threadIdx.x;
  if (d < 25600) { int tap=d>>10, r=d&1023, oc=r>>5, ic=r&31;
    o2[d] = (_Float16)w2[oc*800 + ic*25 + tap]; return; }
  d -= 25600;
  if (d < 51200) { int tap=d>>11, r=d&2047, oc=r>>5, ic=r&31;
    o3[d] = (_Float16)w3[oc*800 + ic*25 + tap]; return; }
  d -= 51200;
  if (d < 102400){ int tap=d>>12, r=d&4095, oc=r>>6, ic=r&63;
    o4[d] = (_Float16)w4[oc*1600 + ic*25 + tap]; return; }
  d -= 102400;
  if (d < 65536) { oc1[d] = (_Float16)cw1[d]; return; }
  d -= 65536;
  if (d < 30720) { oc2[d] = (_Float16)cw2[d]; return; }
  d -= 30720;
  if (d < 65536) { oe1[d] = (_Float16)ew1[d]; return; }
  d -= 65536;
  if (d < 9216)  { oe2[d] = (_Float16)ew2[d]; return; }
}

// ---------------------------------------------------------------------------
// conv1 (VALU) + conv2 (WMMA, 25 taps x K=32) + fused 2x2 maxpool
// one workgroup per image; LDS: act1T 50176B + scratch 9216B = 58KB
// ---------------------------------------------------------------------------
__global__ __launch_bounds__(256) void k_conv12(
    const float* __restrict__ img, const float* __restrict__ w1, const float* __restrict__ b1,
    const _Float16* __restrict__ w2f, const float* __restrict__ b2,
    _Float16* __restrict__ pool1)
{
  __shared__ __attribute__((aligned(16))) _Float16 act1T[784 * 32];
  __shared__ __attribute__((aligned(16))) unsigned int scratch[2304]; // img+w1, then pool enc

  const int b = blockIdx.x;
  const int t = threadIdx.x;
  const int wv = t >> 5;
  const int lane = t & 31;
  const int laneHi = lane >> 4;
  const int ln = lane & 15;

  // stage image (f32) + conv1 weights into the scratch region
  float* shimg = (float*)scratch;
  float* w1s   = (float*)scratch + 1024;
  for (int i = t; i < 1024; i += 256) shimg[i] = img[(size_t)b * 1024 + i];
  for (int i = t; i < 800;  i += 256) w1s[i] = w1[i];
  __syncthreads();

  // conv1: 1->32 ch, 5x5, 32x32 -> 28x28, write act1T[pix][ch] f16
  for (int idx = t; idx < 784 * 32; idx += 256) {
    int ch = idx & 31, pix = idx >> 5;
    int py = pix / 28, px = pix % 28;
    float acc = b1[ch];
    const float* wr = w1s + ch * 25;
    #pragma unroll
    for (int ky = 0; ky < 5; ++ky)
      #pragma unroll
      for (int kx = 0; kx < 5; ++kx)
        acc += shimg[(py + ky) * 32 + (px + kx)] * wr[ky * 5 + kx];
    act1T[pix * 32 + ch] = (_Float16)acc;
  }
  __syncthreads();

  // conv2 via WMMA (M=32 oc, N=576 pix, K=32 ic per tap), two row-halves for LDS budget
  for (int hf = 0; hf < 2; ++hf) {
    for (int i = t; i < 2304; i += 256) scratch[i] = ENC_NEGINF;
    __syncthreads();
    for (int j = wv; j < 36; j += 8) {           // wave-uniform job loop
      int mt = j & 1;
      int nt = hf * 18 + (j >> 1);
      int opix = nt * 16 + ln;
      int py = opix / 24, px = opix % 24;
      f8 acc = {0.f,0.f,0.f,0.f,0.f,0.f,0.f,0.f};
      for (int tap = 0; tap < 25; ++tap) {
        int ky = tap / 5, kx = tap % 5;
        h16 a  = ldfragA(w2f + tap * 1024 + (mt * 16 + ln) * 32, laneHi);
        int spix = (py + ky) * 28 + (px + kx);
        h16 bb = ldfragB(act1T + spix * 32, laneHi);
        acc = WMMA_F16(a, bb, acc);
      }
      int qloc = ((py >> 1) - hf * 6) * 12 + (px >> 1);
      #pragma unroll
      for (int r = 0; r < 8; ++r) {
        int m = mt * 16 + r + 8 * laneHi;
        float v = acc[r] + b2[m];
        atomicMax(&scratch[qloc * 32 + m], fenc(v));
      }
    }
    __syncthreads();
    for (int i = t; i < 2304; i += 256) {
      int q = i >> 5, m = i & 31;
      pool1[(size_t)b * 4608 + (hf * 72 + q) * 32 + m] = (_Float16)fdec(scratch[i]);
    }
    __syncthreads();
  }
}

// ---------------------------------------------------------------------------
// conv3 + conv4 (WMMA) + 2x2 maxpool + torch-order flatten -> emb[256] f16
// ---------------------------------------------------------------------------
__global__ __launch_bounds__(256) void k_conv34(
    const _Float16* __restrict__ pool1,
    const _Float16* __restrict__ w3f, const float* __restrict__ b3,
    const _Float16* __restrict__ w4f, const float* __restrict__ b4,
    _Float16* __restrict__ emb)
{
  __shared__ __attribute__((aligned(16))) _Float16 p1s[144 * 32];
  __shared__ __attribute__((aligned(16))) _Float16 act3T[64 * 64];
  __shared__ unsigned int penc[256];

  const int b = blockIdx.x;
  const int t = threadIdx.x;
  const int wv = t >> 5;
  const int lane = t & 31;
  const int laneHi = lane >> 4;
  const int ln = lane & 15;

  { const unsigned int* src = (const unsigned int*)(pool1 + (size_t)b * 4608);
    unsigned int* dst = (unsigned int*)p1s;
    for (int i = t; i < 2304; i += 256) dst[i] = src[i]; }
  if (t < 256) penc[t] = ENC_NEGINF;
  __syncthreads();

  // conv3: 32->64 ch, 12x12 -> 8x8 (M=64, N=64, K=32 per tap)
  for (int j = wv; j < 16; j += 8) {
    int mt = j >> 2, nt = j & 3;
    int opix = nt * 16 + ln;
    int py = opix >> 3, px = opix & 7;
    f8 acc = {0.f,0.f,0.f,0.f,0.f,0.f,0.f,0.f};
    for (int tap = 0; tap < 25; ++tap) {
      int ky = tap / 5, kx = tap % 5;
      h16 a  = ldfragA(w3f + tap * 2048 + (mt * 16 + ln) * 32, laneHi);
      int spix = (py + ky) * 12 + (px + kx);
      h16 bb = ldfragB(p1s + spix * 32, laneHi);
      acc = WMMA_F16(a, bb, acc);
    }
    int mbase = mt * 16 + 8 * laneHi;
    #pragma unroll
    for (int v = 0; v < 4; ++v) {
      int m0 = mbase + 2 * v;
      union { _Float16 h2[2]; unsigned int u; } pk;
      pk.h2[0] = (_Float16)(acc[2 * v]     + b3[m0]);
      pk.h2[1] = (_Float16)(acc[2 * v + 1] + b3[m0 + 1]);
      *(unsigned int*)&act3T[opix * 64 + m0] = pk.u;
    }
  }
  __syncthreads();

  // conv4: 64->64 ch, 8x8 -> 4x4 (M=64, N=16, K=64 = 2 slices per tap), pool 2x2 -> emb
  if (wv < 4) {                                   // wave-uniform
    int mt = wv;
    int py = ln >> 2, px = ln & 3;
    f8 acc = {0.f,0.f,0.f,0.f,0.f,0.f,0.f,0.f};
    for (int tap = 0; tap < 25; ++tap) {
      int ky = tap / 5, kx = tap % 5;
      int spix = (py + ky) * 8 + (px + kx);
      #pragma unroll
      for (int s = 0; s < 2; ++s) {
        h16 a  = ldfragA(w4f + tap * 4096 + (mt * 16 + ln) * 64 + s * 32, laneHi);
        h16 bb = ldfragB(act3T + spix * 64 + s * 32, laneHi);
        acc = WMMA_F16(a, bb, acc);
      }
    }
    int q = (py >> 1) * 2 + (px >> 1);
    #pragma unroll
    for (int r = 0; r < 8; ++r) {
      int m = mt * 16 + r + 8 * laneHi;
      atomicMax(&penc[m * 4 + q], fenc(acc[r] + b4[m]));
    }
  }
  __syncthreads();
  if (t < 256) emb[(size_t)b * 256 + t] = (_Float16)fdec(penc[t]);
}

// ---------------------------------------------------------------------------
// MLP heads: hidden = relu(emb @ W1^T + b1); out = sigmoid(hidden @ W2^T + b2)
// one 16-row batch tile per block; K=256 = 8 WMMA slices
// ---------------------------------------------------------------------------
__global__ __launch_bounds__(256) void k_heads(
    const _Float16* __restrict__ emb,
    const _Float16* __restrict__ cw1, const float* __restrict__ cb1,
    const _Float16* __restrict__ cw2, const float* __restrict__ cb2,
    const _Float16* __restrict__ ew1, const float* __restrict__ eb1,
    const _Float16* __restrict__ ew2, const float* __restrict__ eb2,
    float* __restrict__ conn, float* __restrict__ ep)
{
  __shared__ __attribute__((aligned(16))) _Float16 hidden[16 * 256];
  const int bt = blockIdx.x;
  const int t = threadIdx.x;
  const int wv = t >> 5;
  const int lane = t & 31;
  const int laneHi = lane >> 4;
  const int ln = lane & 15;

  for (int head = 0; head < 2; ++head) {
    const _Float16* w1p = head ? ew1 : cw1;
    const float*    b1p = head ? eb1 : cb1;
    const _Float16* w2p = head ? ew2 : cw2;
    const float*    b2p = head ? eb2 : cb2;
    const int E = head ? 36 : 120;
    float* outp = head ? ep : conn;

    // layer 1: [16 x 256] = emb_tile[16 x 256] @ W1^T, relu -> LDS f16
    for (int nt = wv; nt < 16; nt += 8) {
      int n = nt * 16 + ln;
      f8 acc = {0.f,0.f,0.f,0.f,0.f,0.f,0.f,0.f};
      for (int ks = 0; ks < 8; ++ks) {
        h16 a  = ldfragA(emb + ((size_t)bt * 16 + ln) * 256 + ks * 32, laneHi);
        h16 bb = ldfragB(w1p + n * 256 + ks * 32, laneHi);
        acc = WMMA_F16(a, bb, acc);
      }
      float bias = b1p[n];
      #pragma unroll
      for (int r = 0; r < 8; ++r) {
        int m = r + 8 * laneHi;
        float v = acc[r] + bias;
        hidden[m * 256 + n] = (_Float16)(v > 0.f ? v : 0.f);
      }
    }
    __syncthreads();

    // layer 2: sigmoid(hidden @ W2^T + b2); pad N up to 16, guard stores
    int nt2 = (E + 15) >> 4;
    if (wv < nt2) {                                // wave-uniform
      int n = wv * 16 + ln;
      bool valid = n < E;
      f8 acc = {0.f,0.f,0.f,0.f,0.f,0.f,0.f,0.f};
      for (int ks = 0; ks < 8; ++ks) {
        h16 a  = ldfragA(hidden + ln * 256 + ks * 32, laneHi);
        h16 bb = ldfragB(w2p + n * 256 + ks * 32, laneHi); // pad rows read inside ws; column-local
        acc = WMMA_F16(a, bb, acc);
      }
      if (valid) {
        float bias = b2p[n];
        #pragma unroll
        for (int r = 0; r < 8; ++r) {
          int m = r + 8 * laneHi;
          float v = acc[r] + bias;
          outp[((size_t)bt * 16 + m) * E + n] = 1.f / (1.f + expf(-v));
        }
      }
    }
    __syncthreads();
  }
}

// ---------------------------------------------------------------------------
// DAMP semiring scan (36 iters, sparse: <=4 predecessors per edge) + endpoint
// ---------------------------------------------------------------------------
__global__ __launch_bounds__(128) void k_paths(
    const float* __restrict__ conn, const float* __restrict__ ep, float* __restrict__ out)
{
  __shared__ int ssrc[120], sdst[120];
  __shared__ float recent[120];
  const int b = blockIdx.x, t = threadIdx.x;
  if (t == 0) {
    int ne = 0;
    const int dxs[4] = {-1, 0, 0, 1};
    const int dys[4] = {0, -1, 1, 0};
    for (int i = 0; i < 6; ++i)
      for (int j = 0; j < 6; ++j)
        for (int d = 0; d < 4; ++d) {
          int x = i + dxs[d], y = j + dys[d];
          if (x >= 0 && x < 6 && y >= 0 && y < 6) { ssrc[ne] = j * 6 + i; sdst[ne] = y * 6 + x; ++ne; }
        }
  }
  __syncthreads();
  const bool valid = t < 120;
  int pred[4]; int np = 0;
  float ce = 0.f, merged = 0.f;
  if (valid) {
    ce = conn[(size_t)b * 120 + t];
    int s0 = ssrc[t];
    for (int e2 = 0; e2 < 120; ++e2)
      if (sdst[e2] == s0 && np < 4) pred[np++] = e2;
    recent[t] = ce;
    merged = ce;
  }
  __syncthreads();
  for (int it = 0; it < 36; ++it) {
    float nv = 0.f;
    if (valid) {
      float s = 0.f;
      #pragma unroll
      for (int p = 0; p < 4; ++p) if (p < np) s += recent[pred[p]];
      s = fminf(fmaxf(s, 0.f), 1.f);
      nv = s * ce;
    }
    __syncthreads();
    if (valid) {
      recent[t] = nv;
      merged = fminf(fmaxf(merged + nv, 0.f), 1.f);
    }
    __syncthreads();
  }
  if (t == 0)
    out[b] = merged * ep[(size_t)b * 36 + ssrc[0]] * ep[(size_t)b * 36 + sdst[0]];
}

// ---------------------------------------------------------------------------
extern "C" void kernel_launch(void* const* d_in, const int* in_sizes, int n_in,
                              void* d_out, int out_size, void* d_ws, size_t ws_size,
                              hipStream_t stream)
{
  (void)in_sizes; (void)n_in; (void)out_size; (void)ws_size;
  const float* image = (const float*)d_in[0];
  const float* c1w = (const float*)d_in[1];  const float* c1b = (const float*)d_in[2];
  const float* c2w = (const float*)d_in[3];  const float* c2b = (const float*)d_in[4];
  const float* c3w = (const float*)d_in[5];  const float* c3b = (const float*)d_in[6];
  const float* c4w = (const float*)d_in[7];  const float* c4b = (const float*)d_in[8];
  const float* epw1 = (const float*)d_in[9];  const float* epb1 = (const float*)d_in[10];
  const float* epw2 = (const float*)d_in[11]; const float* epb2 = (const float*)d_in[12];
  const float* cnw1 = (const float*)d_in[13]; const float* cnb1 = (const float*)d_in[14];
  const float* cnw2 = (const float*)d_in[15]; const float* cnb2 = (const float*)d_in[16];

  char* ws = (char*)d_ws;
  _Float16* w2f   = (_Float16*)(ws + 0);          //  51200 B
  _Float16* w3f   = (_Float16*)(ws + 51200);      // 102400 B
  _Float16* w4f   = (_Float16*)(ws + 153600);     // 204800 B
  _Float16* cw1f  = (_Float16*)(ws + 358400);     // 131072 B
  _Float16* cw2f  = (_Float16*)(ws + 489472);     //  61440 B
  _Float16* ew1f  = (_Float16*)(ws + 550912);     // 131072 B
  _Float16* ew2f  = (_Float16*)(ws + 681984);     //  18432 B
  _Float16* pool1 = (_Float16*)(ws + 700416);     // 1024*144*32*2 = 9437184 B
  _Float16* embf  = (_Float16*)(ws + 10137600);   // 1024*256*2 = 524288 B
  float*    connf = (float*)(ws + 10661888);      // 1024*120*4 = 491520 B
  float*    epf   = (float*)(ws + 11153408);      // 1024*36*4  = 147456 B  (total ~10.8 MB)

  k_cvt   <<<1368, 256, 0, stream>>>(c2w, c3w, c4w, cnw1, cnw2, epw1, epw2,
                                     w2f, w3f, w4f, cw1f, cw2f, ew1f, ew2f);
  k_conv12<<<1024, 256, 0, stream>>>(image, c1w, c1b, w2f, c2b, pool1);
  k_conv34<<<1024, 256, 0, stream>>>(pool1, w3f, c3b, w4f, c4b, embf);
  k_heads <<<64,   256, 0, stream>>>(embf, cw1f, cnb1, cw2f, cnb2, ew1f, epb1, ew2f, epb2,
                                     connf, epf);
  k_paths <<<1024, 128, 0, stream>>>(connf, epf, (float*)d_out);
}
